// NeuralODE_53575422051098
// MI455X (gfx1250) — matrix operand order repaired and checked by reading, not compiled, not run on hardware
//
#include <hip/hip_runtime.h>

// ---------------- problem constants ----------------
#define DIMN   64
#define WIDTH  256
#define TSTEPS 256
#define BATCH  4096
#define ROWS_PER_WG 32          // 4096/32 = 128 workgroups (~1 per WGP)
#define HB_STRIDE 256           // halves
#define HIN_STRIDE 80           // 64 + 16 halves pad

// Tsit5 coefficients
static __device__ constexpr float A21f = 0.161f;
static __device__ constexpr float A31f = -0.008480655492356989f, A32f = 0.335480655492357f;
static __device__ constexpr float A41f = 2.8971530571054935f,  A42f = -6.359448489975075f,  A43f = 4.3622954328695815f;
static __device__ constexpr float A51f = 5.325864828439257f,   A52f = -11.748883564062828f, A53f = 7.4955393428898365f, A54f = -0.09249506636175525f;
static __device__ constexpr float A61f = 5.86145544294642f,    A62f = -12.92096931784711f,  A63f = 8.159367898576159f,  A64f = -0.071584973281401f, A65f = -0.028269050394068383f;
static __device__ constexpr float B1f = 0.09646076681806523f,  B2f = 0.01f, B3f = 0.4798896504144996f;
static __device__ constexpr float B4f = 1.379008574103742f,    B5f = -3.290069515436081f,   B6f = 2.324710524099774f;

typedef _Float16 v16h __attribute__((ext_vector_type(16)));
typedef _Float16 v8h  __attribute__((ext_vector_type(8)));
typedef float    v8f  __attribute__((ext_vector_type(8)));

#define WMMA_F16(a, b, c) \
    __builtin_amdgcn_wmma_f32_16x16x32_f16(false, (a), false, (b), (short)0, (c), false, false)

#if __has_builtin(__builtin_amdgcn_tanhf)
#define FAST_TANH(x) __builtin_amdgcn_tanhf(x)
#else
#define FAST_TANH(x) tanhf(x)
#endif

// A-operand (16x32 f16): lane L (r = L%16, kh = L/16) holds row r,
// K chunks [8*kh .. 8*kh+7] and [16+8*kh .. 23+8*kh]  (ISA 7.12.2 table)
static __device__ __forceinline__ v16h ldA(const _Float16* rowk0, int kh) {
    const v8h lo = *(const v8h*)(rowk0 + kh * 8);
    const v8h hi = *(const v8h*)(rowk0 + 16 + kh * 8);
    return __builtin_shufflevector(lo, hi, 0,1,2,3,4,5,6,7,8,9,10,11,12,13,14,15);
}
// B-operand (32x16 f16): lane L holds column n = L%16, K chunks [8*kh .. 8*kh+7] and [16+8*kh .. 23+8*kh]
static __device__ __forceinline__ v16h ldB(const _Float16* colrow, int kh) {
    const v8h lo = *(const v8h*)(colrow + kh * 8);
    const v8h hi = *(const v8h*)(colrow + 16 + kh * 8);
    return __builtin_shufflevector(lo, hi, 0,1,2,3,4,5,6,7,8,9,10,11,12,13,14,15);
}

// ---------------- weight preprocessing: fp32 row-major -> f16 [N][K] ----------------
// d_ws layout (halves): [0,131072) WhT (2 x 256x256), [131072,147456) WinT (256x64),
//                       [147456,163840) WoutT (64x256)
__global__ void prep_weights(const float* __restrict__ W_in,
                             const float* __restrict__ W_h,
                             const float* __restrict__ W_out,
                             _Float16* __restrict__ wsT) {
    int idx = blockIdx.x * 256 + threadIdx.x;
    if (idx < 131072) {                         // WhT[l][n][k] = W_h[l][k][n]
        int l = idx >> 16, rr = idx & 65535;
        int n = rr >> 8, k = rr & 255;
        wsT[idx] = (_Float16)W_h[(l * 256 + k) * 256 + n];
    } else if (idx < 147456) {                  // WinT[n][k] = W_in[k][n]
        int j = idx - 131072;
        int n = j >> 6, k = j & 63;
        wsT[idx] = (_Float16)W_in[k * 256 + n];
    } else if (idx < 163840) {                  // WoutT[n][k] = W_out[k][n]
        int j = idx - 147456;
        int n = j >> 8, k = j & 255;
        wsT[idx] = (_Float16)W_out[k * 64 + n];
    }
}

// ---------------- one MLP vector-field evaluation (per workgroup, 32 rows) ----------
static __device__ __forceinline__ void run_mlp(
    const _Float16* __restrict__ sWh,   // LDS: [2][256][256] transposed f16
    const _Float16* __restrict__ gWinT, // global: [256][64] transposed f16 (reg-hoisted)
    const _Float16* __restrict__ gWoutT,// global: [64][256] transposed f16 (reg-hoisted)
    const float* __restrict__ b_in, const float* __restrict__ b_h,
    const float* __restrict__ b_out,
    _Float16* __restrict__ hin,         // LDS in:  [32][HIN_STRIDE] f16
    _Float16* __restrict__ hb0,         // LDS tmp: [32][HB_STRIDE]  f16
    _Float16* __restrict__ hb1,         // LDS tmp: [32][HB_STRIDE]  f16
    float* __restrict__ kbuf,           // LDS out: [32][64] f32
    int wave, int r, int kh)
{
    // ---- layer 0: [32x64] @ WinT -> hb1 [32x256], tanh ----
    {
        const int n0 = wave * 32;
        v8f c00, c01, c10, c11;
        const float bb0 = b_in[n0 + r], bb1 = b_in[n0 + 16 + r];
#pragma unroll
        for (int v = 0; v < 8; ++v) { c00[v] = bb0; c01[v] = bb1; c10[v] = bb0; c11[v] = bb1; }
#pragma unroll
        for (int k0 = 0; k0 < DIMN; k0 += 32) {
            v16h a0 = ldA(hin + r * HIN_STRIDE + k0, kh);
            v16h a1 = ldA(hin + (16 + r) * HIN_STRIDE + k0, kh);
            v16h b0 = ldB(gWinT + (n0 + r) * DIMN + k0, kh);
            v16h b1 = ldB(gWinT + (n0 + 16 + r) * DIMN + k0, kh);
            c00 = WMMA_F16(a0, b0, c00);
            c01 = WMMA_F16(a0, b1, c01);
            c10 = WMMA_F16(a1, b0, c10);
            c11 = WMMA_F16(a1, b1, c11);
        }
#pragma unroll
        for (int v = 0; v < 8; ++v) {
            const int m = kh * 8 + v;
            hb1[m * HB_STRIDE + n0 + r]             = (_Float16)FAST_TANH(c00[v]);
            hb1[m * HB_STRIDE + n0 + 16 + r]        = (_Float16)FAST_TANH(c01[v]);
            hb1[(16 + m) * HB_STRIDE + n0 + r]      = (_Float16)FAST_TANH(c10[v]);
            hb1[(16 + m) * HB_STRIDE + n0 + 16 + r] = (_Float16)FAST_TANH(c11[v]);
        }
    }
    __syncthreads();

    // ---- hidden layers: hb1 -> hb0 -> hb1, tanh ----
#pragma unroll
    for (int l = 0; l < 2; ++l) {
        const _Float16* hi = l ? hb0 : hb1;
        _Float16*       ho = l ? hb1 : hb0;
        const _Float16* W  = sWh + l * (WIDTH * WIDTH);
        const float*    bh = b_h + l * WIDTH;
        const int n0 = wave * 32;
        v8f c00, c01, c10, c11;
        const float bb0 = bh[n0 + r], bb1 = bh[n0 + 16 + r];
#pragma unroll
        for (int v = 0; v < 8; ++v) { c00[v] = bb0; c01[v] = bb1; c10[v] = bb0; c11[v] = bb1; }
#pragma unroll
        for (int k0 = 0; k0 < WIDTH; k0 += 32) {
            v16h a0 = ldA(hi + r * HB_STRIDE + k0, kh);
            v16h a1 = ldA(hi + (16 + r) * HB_STRIDE + k0, kh);
            v16h b0 = ldB(W + (n0 + r) * WIDTH + k0, kh);
            v16h b1 = ldB(W + (n0 + 16 + r) * WIDTH + k0, kh);
            c00 = WMMA_F16(a0, b0, c00);
            c01 = WMMA_F16(a0, b1, c01);
            c10 = WMMA_F16(a1, b0, c10);
            c11 = WMMA_F16(a1, b1, c11);
        }
#pragma unroll
        for (int v = 0; v < 8; ++v) {
            const int m = kh * 8 + v;
            ho[m * HB_STRIDE + n0 + r]             = (_Float16)FAST_TANH(c00[v]);
            ho[m * HB_STRIDE + n0 + 16 + r]        = (_Float16)FAST_TANH(c01[v]);
            ho[(16 + m) * HB_STRIDE + n0 + r]      = (_Float16)FAST_TANH(c10[v]);
            ho[(16 + m) * HB_STRIDE + n0 + 16 + r] = (_Float16)FAST_TANH(c11[v]);
        }
        __syncthreads();
    }

    // ---- output layer: [32x256] @ WoutT -> kbuf [32x64] f32 (one tile per wave) ----
    {
        const int mt = wave >> 2;       // M-tile 0..1
        const int nt = wave & 3;        // N-tile 0..3
        v8f c0;
        const float bb = b_out[nt * 16 + r];
#pragma unroll
        for (int v = 0; v < 8; ++v) c0[v] = bb;
#pragma unroll
        for (int k0 = 0; k0 < WIDTH; k0 += 32) {
            v16h a  = ldA(hb1 + (mt * 16 + r) * HB_STRIDE + k0, kh);
            v16h b0 = ldB(gWoutT + (nt * 16 + r) * WIDTH + k0, kh);
            c0 = WMMA_F16(a, b0, c0);
        }
#pragma unroll
        for (int v = 0; v < 8; ++v)
            kbuf[(mt * 16 + kh * 8 + v) * 64 + nt * 16 + r] = c0[v];
    }
    __syncthreads();
}

// ---------------- main persistent kernel: full Tsit5 trajectory ------------------
__launch_bounds__(256, 1)
__global__ void ode_tsit5_kernel(const float* __restrict__ ts,
                                 const float* __restrict__ y0,
                                 const float* __restrict__ b_in,
                                 const float* __restrict__ b_h,
                                 const float* __restrict__ b_out,
                                 const _Float16* __restrict__ wsT,
                                 float* __restrict__ out)
{
    __shared__ __align__(32) _Float16 sWh[2 * WIDTH * WIDTH];            // 256 KB
    __shared__ __align__(32) _Float16 hin[ROWS_PER_WG * HIN_STRIDE];     // 5 KB
    __shared__ __align__(32) _Float16 hb0[ROWS_PER_WG * HB_STRIDE];      // 16 KB
    __shared__ __align__(32) _Float16 hb1[ROWS_PER_WG * HB_STRIDE];      // 16 KB
    __shared__ __align__(16) float    kbuf[ROWS_PER_WG * DIMN];          // 8 KB

    const int tid  = threadIdx.x;
    const int wave = tid >> 5;
    const int lane = tid & 31;
    const int r    = lane & 15;
    const int kh   = lane >> 4;

    // ---- stage W_h into LDS via async global->LDS DMA (ASYNCcnt path) ----
    {
        const uint32_t lds_base = (uint32_t)(uintptr_t)(&sWh[0]);
        const uint64_t gbase    = (uint64_t)(uintptr_t)wsT;
        for (int i = tid; i < 16384; i += 256) {          // 16384 x 16B = 256 KB
            uint32_t ldsa = lds_base + (uint32_t)i * 16u;
            uint64_t ga   = gbase + (uint64_t)i * 16u;
            asm volatile("global_load_async_to_lds_b128 %0, %1, off"
                         :: "v"(ldsa), "v"(ga) : "memory");
        }
        asm volatile("s_wait_asynccnt 0" ::: "memory");
    }
    const _Float16* gWinT  = wsT + 131072;
    const _Float16* gWoutT = wsT + 147456;

    // this workgroup's 32 batch rows; thread owns 8 contiguous state elements
    const int gvec = blockIdx.x * 512 + tid * 2;   // float4 index into [B*DIM]
    float4 yl0 = ((const float4*)y0)[gvec];
    float4 yl1 = ((const float4*)y0)[gvec + 1];
    float yv[8] = {yl0.x, yl0.y, yl0.z, yl0.w, yl1.x, yl1.y, yl1.z, yl1.w};
    ((float4*)out)[gvec]     = yl0;                // out[0] = y0
    ((float4*)out)[gvec + 1] = yl1;
    float ks[6][8];

    const int m_own = tid >> 3;                    // row 0..31
    const int c_own = (tid & 7) * 8;               // col 0..56
    _Float16* hin_own = hin + m_own * HIN_STRIDE + c_own;

    __syncthreads();

#define BUILD_HIN(EXPR)                                   \
    {                                                     \
        v8h hv;                                           \
        _Pragma("unroll")                                 \
        for (int q = 0; q < 8; ++q) {                     \
            const float zz = (EXPR);                      \
            hv[q] = (_Float16)zz;                         \
        }                                                 \
        *(v8h*)hin_own = hv;                              \
    }                                                     \
    __syncthreads();

#define RUN_STAGE(sidx)                                                        \
    run_mlp(sWh, gWinT, gWoutT, b_in, b_h, b_out, hin, hb0, hb1, kbuf,         \
            wave, r, kh);                                                      \
    _Pragma("unroll")                                                          \
    for (int q = 0; q < 8; ++q) ks[sidx][q] = kbuf[tid * 8 + q];

    for (int t = 0; t < TSTEPS - 1; ++t) {
        const float hdt = ts[t + 1] - ts[t];

        BUILD_HIN(yv[q]);
        RUN_STAGE(0);
        BUILD_HIN(yv[q] + hdt * (A21f * ks[0][q]));
        RUN_STAGE(1);
        BUILD_HIN(yv[q] + hdt * (A31f * ks[0][q] + A32f * ks[1][q]));
        RUN_STAGE(2);
        BUILD_HIN(yv[q] + hdt * (A41f * ks[0][q] + A42f * ks[1][q] + A43f * ks[2][q]));
        RUN_STAGE(3);
        BUILD_HIN(yv[q] + hdt * (A51f * ks[0][q] + A52f * ks[1][q] + A53f * ks[2][q] + A54f * ks[3][q]));
        RUN_STAGE(4);
        BUILD_HIN(yv[q] + hdt * (A61f * ks[0][q] + A62f * ks[1][q] + A63f * ks[2][q] + A64f * ks[3][q] + A65f * ks[4][q]));
        RUN_STAGE(5);

#pragma unroll
        for (int q = 0; q < 8; ++q)
            yv[q] += hdt * (B1f * ks[0][q] + B2f * ks[1][q] + B3f * ks[2][q] +
                            B4f * ks[3][q] + B5f * ks[4][q] + B6f * ks[5][q]);

        float4 o0, o1;
        o0.x = yv[0]; o0.y = yv[1]; o0.z = yv[2]; o0.w = yv[3];
        o1.x = yv[4]; o1.y = yv[5]; o1.z = yv[6]; o1.w = yv[7];
        const size_t obase = (size_t)(t + 1) * (BATCH * DIMN / 4) + gvec;
        ((float4*)out)[obase]     = o0;
        ((float4*)out)[obase + 1] = o1;
    }
#undef BUILD_HIN
#undef RUN_STAGE
}

// ---------------- launcher ----------------
extern "C" void kernel_launch(void* const* d_in, const int* in_sizes, int n_in,
                              void* d_out, int out_size, void* d_ws, size_t ws_size,
                              hipStream_t stream) {
    const float* ts    = (const float*)d_in[0];
    const float* y0    = (const float*)d_in[1];
    const float* W_in  = (const float*)d_in[2];
    const float* b_in  = (const float*)d_in[3];
    const float* W_h   = (const float*)d_in[4];
    const float* b_h   = (const float*)d_in[5];
    const float* W_out = (const float*)d_in[6];
    const float* b_out = (const float*)d_in[7];
    float* out = (float*)d_out;
    _Float16* wsT = (_Float16*)d_ws;

    prep_weights<<<640, 256, 0, stream>>>(W_in, W_h, W_out, wsT);
    ode_tsit5_kernel<<<BATCH / ROWS_PER_WG, 256, 0, stream>>>(
        ts, y0, b_in, b_h, b_out, wsT, out);
}
